// MultiHeadAttention_10780367913504
// MI455X (gfx1250) — matrix operand order repaired and checked by reading, not compiled, not run on hardware
//
#include <hip/hip_runtime.h>
#include <hip/hip_bf16.h>

typedef _Float16 half_t;
typedef __attribute__((ext_vector_type(16))) _Float16 v16h;
typedef __attribute__((ext_vector_type(8)))  _Float16 v8h;
typedef __attribute__((ext_vector_type(8)))  float    v8f;

#define D_MODEL   1024
#define NUM_HEADS 16
#define D_K       64
#define BATCH     2
#define SEQ       2048
#define ROWS      (BATCH * SEQ)        // 4096
#define LDS_PITCH 40                   // halves per row, padded (80B, 16B aligned)

// ---------------------------------------------------------------- fp32 -> f16
__global__ void cvt_f32_f16(const float* __restrict__ src,
                            half_t* __restrict__ dst, int n) {
  int i = blockIdx.x * blockDim.x + threadIdx.x;
  int stride = gridDim.x * blockDim.x;
  for (; i < n; i += stride) dst[i] = (half_t)src[i];
}

// ---------------------------------------------------------------- GEMM: Y = X @ W^T + b
// X is [4096 x 1024] (f32 for MODE 0/1, f16 for MODE 2); W is [1024 x 1024] f16 row-major.
// Per wave: 16x64 output strip, K-loop step 32 (v_wmma_f32_16x16x32_f16), A-fragment reuse x4.
// MODE 0: f16 out, head-split     out[((b*H+h)*S + s)*64 + d]   (q, k)
// MODE 1: f16 out, head-split^T   out[((b*H+h)*64 + d)*S + s]   (v transposed for P@V)
// MODE 2: f32 out, row-major      out[m*1024 + n]               (final projection)
template <int MODE>
__global__ __launch_bounds__(256) void gemm_kernel(
    const float* __restrict__ Xf,
    const half_t* __restrict__ Xh,
    const half_t* __restrict__ Wh,
    const float* __restrict__ bias,
    half_t* __restrict__ outh,
    float* __restrict__ outf) {
  const int lane = threadIdx.x & 31;
  const int wave = threadIdx.x >> 5;
  const int hf   = lane >> 4;   // 0|1 : half-wave
  const int l16  = lane & 15;

  const int m0 = (blockIdx.y * 8 + wave) * 16;   // row tile
  const int n0 = blockIdx.x * 64;                // col strip
  const int m  = m0 + l16;

  // straight-line pointer induction (no in-loop mode branch: MODE is a template arg)
  const float*  ap32 = (MODE == 2) ? nullptr : (Xf + (size_t)m * D_MODEL + hf * 16);
  const half_t* ap16 = (MODE == 2) ? (Xh + (size_t)m * D_MODEL + hf * 16) : nullptr;
  const half_t* bp[4];
#pragma unroll
  for (int f = 0; f < 4; ++f)
    bp[f] = Wh + (size_t)(n0 + f * 16 + l16) * D_MODEL + hf * 16;

  v8f acc[4] = {};

  for (int k0 = 0; k0 < D_MODEL; k0 += 32) {
    // A fragment: row m; lane<16 gets K 0-15; lane>=16 gets K 16-31
    v16h a;
    if (MODE == 2) {
      v8h lo = *(const v8h*)(ap16);
      v8h hi = *(const v8h*)(ap16 + 8);
      ap16 += 32;
#pragma unroll
      for (int j = 0; j < 8; ++j) { a[j] = lo[j]; a[8 + j] = hi[j]; }
    } else {
      v8f lo = *(const v8f*)(ap32);
      v8f hi = *(const v8f*)(ap32 + 8);
      ap32 += 32;
#pragma unroll
      for (int j = 0; j < 8; ++j) { a[j] = (half_t)lo[j]; a[8 + j] = (half_t)hi[j]; }
    }
#pragma unroll
    for (int f = 0; f < 4; ++f) {
      // B fragment: B[k][n] = W[n][k]; N = l16, K = 16*hf + j  -> one 32B contiguous load
      v16h bfrag = *(const v16h*)(bp[f]);
      bp[f] += 32;
      acc[f] = __builtin_amdgcn_wmma_f32_16x16x32_f16(
          false, a, false, bfrag, (short)0, acc[f], false, false);
    }
  }

  // epilogue: bias + store
#pragma unroll
  for (int f = 0; f < 4; ++f) {
    const int n  = n0 + f * 16 + l16;
    const float bv = bias[n];
    const int h = n >> 6;    // head
    const int d = n & 63;
#pragma unroll
    for (int r = 0; r < 8; ++r) {
      const int mm = m0 + r + hf * 8;   // C layout: M = r + 8*half
      const float y = acc[f][r] + bv;
      if (MODE == 2) {
        outf[(size_t)mm * D_MODEL + n] = y;
      } else {
        const int b  = mm >> 11;        // mm / SEQ
        const int s  = mm & (SEQ - 1);
        const int bh = b * NUM_HEADS + h;
        const size_t idx = (MODE == 0)
            ? ((size_t)bh * SEQ + s) * D_K + d
            : ((size_t)bh * D_K + d) * SEQ + s;
        outh[idx] = (half_t)y;
      }
    }
  }
}

// ---------------------------------------------------------------- flash attention
// One wave per 16-query tile; key chunks of 32 (causal-truncated); online softmax.
__global__ __launch_bounds__(256) void attn_kernel(
    const half_t* __restrict__ qh,
    const half_t* __restrict__ kh,
    const half_t* __restrict__ vTh,
    half_t* __restrict__ attnh) {
  __shared__ half_t lds[8][16 * LDS_PITCH];   // per-wave P-tile staging (C -> A relayout)

  const int lane = threadIdx.x & 31;
  const int wave = threadIdx.x >> 5;
  const int hf   = lane >> 4;
  const int l16  = lane & 15;

  const int gt = blockIdx.x * 8 + wave;     // 0..255
  const int q0 = (gt & 127) * 16;           // query tile start (S/16 = 128)
  const int bh = gt >> 7;                   // b*NUM_HEADS + h

  const half_t* qbase = qh  + ((size_t)bh * SEQ + q0) * D_K;
  const half_t* kbase = kh  + (size_t)bh * SEQ * D_K;
  const half_t* vbase = vTh + (size_t)bh * D_K * SEQ;
  half_t* plds = &lds[wave][0];

  // q A-fragments for d = 0..31 and 32..63 (fixed for the whole key loop)
  v16h qa[2];
  {
    const half_t* rp = qbase + (size_t)l16 * D_K;
#pragma unroll
    for (int t = 0; t < 2; ++t) {
      v8h lo = *(const v8h*)(rp + t * 32 + hf * 16);
      v8h hi = *(const v8h*)(rp + t * 32 + hf * 16 + 8);
#pragma unroll
      for (int j = 0; j < 8; ++j) { qa[t][j] = lo[j]; qa[t][8 + j] = hi[j]; }
    }
  }

  float m_i[8], l_i[8];
  v8f O[4] = {};
#pragma unroll
  for (int r = 0; r < 8; ++r) { m_i[r] = -3.0e38f; l_i[r] = 0.0f; }

  const float scale = 0.125f;               // 1/sqrt(64)
  const int nchunks = (q0 >> 5) + 1;        // keys <= q0+15 needed

  for (int c = 0; c < nchunks; ++c) {
    const int key0 = c * 32;

    // scores: two 16x16 tiles, K=64 over d (2 WMMAs each)
    v8f sc[2] = {};
#pragma unroll
    for (int t = 0; t < 2; ++t) {
      const half_t* kp = kbase + (size_t)(key0 + t * 16 + l16) * D_K;
#pragma unroll
      for (int kk = 0; kk < 2; ++kk) {
        v16h bfrag = *(const v16h*)(kp + kk * 32 + hf * 16);
        sc[t] = __builtin_amdgcn_wmma_f32_16x16x32_f16(
            false, qa[kk], false, bfrag, (short)0, sc[t], false, false);
      }
    }

    // scale + causal mask + online softmax (row = r + 8*hf, cols across 16 lanes)
    float alpha[8];
#pragma unroll
    for (int r = 0; r < 8; ++r) {
      const int qrow = q0 + r + 8 * hf;
      float s0 = sc[0][r] * scale;
      float s1 = sc[1][r] * scale;
      if (key0 + l16 > qrow)      s0 -= 1.0e9f;   // additive mask like reference
      if (key0 + 16 + l16 > qrow) s1 -= 1.0e9f;
      float mx = fmaxf(s0, s1);
#pragma unroll
      for (int d = 1; d < 16; d <<= 1) mx = fmaxf(mx, __shfl_xor(mx, d, 32));
      const float mnew = fmaxf(m_i[r], mx);
      alpha[r] = __expf(m_i[r] - mnew);
      const float p0 = __expf(s0 - mnew);
      const float p1 = __expf(s1 - mnew);
      float rs = p0 + p1;
#pragma unroll
      for (int d = 1; d < 16; d <<= 1) rs += __shfl_xor(rs, d, 32);
      l_i[r] = l_i[r] * alpha[r] + rs;
      m_i[r] = mnew;
      // stage P (f16) row-major for A-fragment reload
      plds[(r + 8 * hf) * LDS_PITCH + l16]      = (half_t)p0;
      plds[(r + 8 * hf) * LDS_PITCH + 16 + l16] = (half_t)p1;
    }

    // rescale accumulator
#pragma unroll
    for (int f = 0; f < 4; ++f)
#pragma unroll
      for (int r = 0; r < 8; ++r) O[f][r] *= alpha[r];

    // per-wave LDS fence (chunk counts differ per wave -> no block barrier)
    asm volatile("s_wait_dscnt 0" ::: "memory");

    // reload P as 16x32 A-fragment
    v16h pa;
    {
      const half_t* rp = plds + l16 * LDS_PITCH;
      v8h lo = *(const v8h*)(rp + hf * 16);
      v8h hi = *(const v8h*)(rp + hf * 16 + 8);
#pragma unroll
      for (int j = 0; j < 8; ++j) { pa[j] = lo[j]; pa[8 + j] = hi[j]; }
    }

    // O += P @ V : B[kk][d] = vT[d][key0+kk] -> contiguous 32B loads
#pragma unroll
    for (int f = 0; f < 4; ++f) {
      const half_t* vp = vbase + (size_t)(f * 16 + l16) * SEQ + key0 + hf * 16;
      v16h bfrag = *(const v16h*)vp;
      O[f] = __builtin_amdgcn_wmma_f32_16x16x32_f16(
          false, pa, false, bfrag, (short)0, O[f], false, false);
    }
  }

  // epilogue: normalize and store f16 to attn buffer [4096 x 1024]
  const int bb = bh >> 4;
  const int h  = bh & 15;
#pragma unroll
  for (int f = 0; f < 4; ++f)
#pragma unroll
    for (int r = 0; r < 8; ++r) {
      const int qrow = q0 + r + 8 * hf;
      const int col  = h * 64 + f * 16 + l16;
      attnh[((size_t)(bb * SEQ + qrow)) * D_MODEL + col] =
          (half_t)(O[f][r] / l_i[r]);
    }
}

// ---------------------------------------------------------------- launcher
extern "C" void kernel_launch(void* const* d_in, const int* in_sizes, int n_in,
                              void* d_out, int out_size, void* d_ws, size_t ws_size,
                              hipStream_t stream) {
  (void)in_sizes; (void)n_in; (void)out_size; (void)ws_size;
  const float* Q  = (const float*)d_in[0];
  const float* K  = (const float*)d_in[1];
  const float* V  = (const float*)d_in[2];
  /* d_in[3] = mask: causal triu(k=1), applied analytically in-kernel */
  const float* Wq = (const float*)d_in[4];
  const float* bq = (const float*)d_in[5];
  const float* Wk = (const float*)d_in[6];
  const float* bk = (const float*)d_in[7];
  const float* Wv = (const float*)d_in[8];
  const float* bv = (const float*)d_in[9];
  const float* Wo = (const float*)d_in[10];
  const float* bo = (const float*)d_in[11];
  float* out = (float*)d_out;

  char* ws = (char*)d_ws;
  const size_t MB = 1024 * 1024;
  half_t* Wqh = (half_t*)(ws + 0 * MB);
  half_t* Wkh = (half_t*)(ws + 2 * MB);
  half_t* Wvh = (half_t*)(ws + 4 * MB);
  half_t* Woh = (half_t*)(ws + 6 * MB);
  half_t* qh  = (half_t*)(ws + 8 * MB);   // [B,H,S,64]
  half_t* kh  = (half_t*)(ws + 16 * MB);  // [B,H,S,64]
  half_t* vTh = (half_t*)(ws + 24 * MB);  // [B,H,64,S]
  half_t* ah  = (half_t*)(ws + 32 * MB);  // [4096,1024]

  const int nw = D_MODEL * D_MODEL;
  cvt_f32_f16<<<512, 256, 0, stream>>>(Wq, Wqh, nw);
  cvt_f32_f16<<<512, 256, 0, stream>>>(Wk, Wkh, nw);
  cvt_f32_f16<<<512, 256, 0, stream>>>(Wv, Wvh, nw);
  cvt_f32_f16<<<512, 256, 0, stream>>>(Wo, Woh, nw);

  dim3 gblock(256);
  dim3 ggrid(D_MODEL / 64, ROWS / 16 / 8);   // (16, 32)
  gemm_kernel<0><<<ggrid, gblock, 0, stream>>>(Q, nullptr, Wqh, bq, qh,  nullptr);
  gemm_kernel<0><<<ggrid, gblock, 0, stream>>>(K, nullptr, Wkh, bk, kh,  nullptr);
  gemm_kernel<1><<<ggrid, gblock, 0, stream>>>(V, nullptr, Wvh, bv, vTh, nullptr);

  attn_kernel<<<ROWS / 16 / 8, 256, 0, stream>>>(qh, kh, vTh, ah);

  gemm_kernel<2><<<ggrid, gblock, 0, stream>>>(nullptr, ah, Woh, bo, nullptr, out);
}
